// BackflowNet_10110353015415
// MI455X (gfx1250) — matrix operand order repaired and checked by reading, not compiled, not run on hardware
//
#include <hip/hip_runtime.h>
#include <hip/hip_fp16.h>
#include <math.h>

#define BATCH 512
#define NPART 64
#define DIM   3
#define H     128
#define LDS_STRIDE 132   // 16 rows * 132 dwords: +4 dword pad to spread LDS banks

typedef __attribute__((ext_vector_type(16))) _Float16 v16h_t;
typedef __attribute__((ext_vector_type(8)))  float    v8f_t;

union V16H { v16h_t v; _Float16 h[16]; unsigned int u[8]; };
union PK   { _Float16 h[2]; unsigned int u; };

// ---- raw-TRANS-op fast math (v_exp_f32 / v_rcp_f32 / v_sqrt_f32 / v_tanh_f32)
__device__ __forceinline__ float fast_exp(float x) {
  return __builtin_amdgcn_exp2f(x * 1.44269504088896340736f);
}
__device__ __forceinline__ float fast_rcp(float x) {
  return __builtin_amdgcn_rcpf(x);
}
__device__ __forceinline__ float fast_sqrt(float x) {
  return __builtin_amdgcn_sqrtf(x);
}
__device__ __forceinline__ float silu_f(float x) {
  return x * fast_rcp(1.0f + fast_exp(-x));
}
__device__ __forceinline__ float fast_tanh(float x) {
#if __has_builtin(__builtin_amdgcn_tanhf)
  return __builtin_amdgcn_tanhf(x);
#else
  // tanh(x) = 1 - 2/(exp(2x)+1)
  float e = __builtin_amdgcn_exp2f(x * 2.88539008177792681472f);
  return 1.0f - 2.0f * __builtin_amdgcn_rcpf(e + 1.0f);
#endif
}

// ---------------------------------------------------------------------------
// Pack a (fan_in x fan_out) row-major f32 weight into the WMMA f16 B-operand
// layout, one K=32 chunk x 16-col tile at a time.
//   dst[(((chunk*n_ct + c)*32 + lane)*8 + v)]:
//     lane<16 : col = 16c+lane,    K = 32*chunk + 2v + p (v<4), 32*chunk + 8 + 2v + p (v>=4)
//     lane>=16: col = 16c+lane-16, K = 32*chunk + 8 + 2v + p (v<4), 32*chunk + 16 + 2v + p (v>=4)
// Out-of-range K/cols are zero-padded (K=11 layer, K=3 x-rows, 3-col output).
// ---------------------------------------------------------------------------
__global__ void prep_weights(const float* __restrict__ W, unsigned int* __restrict__ dst,
                             int row_offset, int row_count, int ld, int ncols_valid,
                             int n_ct, int n_chunks) {
  int tid = blockIdx.x * blockDim.x + threadIdx.x;
  int total = n_chunks * n_ct * 256;
  if (tid >= total) return;
  int v     = tid & 7;
  int lane  = (tid >> 3) & 31;
  int c     = (tid >> 8) % n_ct;
  int chunk = tid / (n_ct * 256);
  int col = c * 16 + (lane & 15);
  int k   = chunk * 32 + ((lane < 16) ? 0 : 8) + 2 * v + ((v >= 4) ? 8 : 0);
  float f0 = 0.f, f1 = 0.f;
  if (col < ncols_valid) {
    if (k     < row_count) f0 = W[(size_t)(row_offset + k    ) * ld + col];
    if (k + 1 < row_count) f1 = W[(size_t)(row_offset + k + 1) * ld + col];
  }
  PK p; p.h[0] = (_Float16)f0; p.h[1] = (_Float16)f1;
  dst[tid] = p.u;
}

// ---------------------------------------------------------------------------
// Phase 1: s[b,i,:] = sum_{j != i} silu( features(b,i,j) @ phi_w0 + phi_b0 )
// One wave per (b,i). Features K=11 padded to one K=32 chunk.
// 32 x v_wmma_f32_16x16x32_f16 per wave.
// ---------------------------------------------------------------------------
__global__ __launch_bounds__(256) void phase1(
    const float* __restrict__ x, const unsigned int* __restrict__ w0p,
    const float* __restrict__ b0, float* __restrict__ s) {
  int wave = threadIdx.x >> 5;
  int lane = threadIdx.x & 31;
  int wid  = blockIdx.x * 8 + wave;       // flat (b,i)
  int b = wid >> 6;
  int i = wid & 63;
  int lrow = lane & 15;
  bool lo = lane < 16;

  const float* xb = x + (size_t)b * NPART * DIM;
  float xi0 = xb[i*3+0], xi1 = xb[i*3+1], xi2 = xb[i*3+2];

  // constant B operands for the single K-chunk of phi layer 0 (8 col tiles)
  V16H Bop[8];
#pragma unroll
  for (int c = 0; c < 8; ++c) {
    const unsigned int* p = w0p + ((size_t)(c * 32 + lane)) * 8;
#pragma unroll
    for (int vv = 0; vv < 8; ++vv) Bop[c].u[vv] = p[vv];
  }
  float bias[8], sum[8];
#pragma unroll
  for (int c = 0; c < 8; ++c) { bias[c] = b0[c*16 + lrow]; sum[c] = 0.f; }

#pragma unroll
  for (int jt = 0; jt < 4; ++jt) {
    int j = jt * 16 + lrow;               // this lane's A-row (one j per row)
    float xj0 = xb[j*3+0], xj1 = xb[j*3+1], xj2 = xb[j*3+2];
    float d0 = xi0-xj0, d1 = xi1-xj1, d2 = xi2-xj2;
    float rr = d0*d0 + d1*d1 + d2*d2;
    float rn = fast_sqrt(rr + 1e-12f);

    // A-operand layout (16-bit A 16x32): lanes 0-15 hold K=0..7 (+16..23),
    // lanes 16-31 hold K=8..15 (+24..31). Features: [xi(3),xj(3),r(3),|r|,r2]
    V16H A;
#pragma unroll
    for (int e = 0; e < 16; ++e) A.h[e] = (_Float16)0.0f;
    if (lo) {
      A.h[0]=(_Float16)xi0; A.h[1]=(_Float16)xi1; A.h[2]=(_Float16)xi2;
      A.h[3]=(_Float16)xj0; A.h[4]=(_Float16)xj1; A.h[5]=(_Float16)xj2;
      A.h[6]=(_Float16)d0;  A.h[7]=(_Float16)d1;
    } else {
      A.h[0]=(_Float16)d2;  A.h[1]=(_Float16)rn; A.h[2]=(_Float16)rr;
    }

#pragma unroll
    for (int c = 0; c < 8; ++c) {
      v8f_t acc;
#pragma unroll
      for (int r = 0; r < 8; ++r) acc[r] = bias[c];
      acc = __builtin_amdgcn_wmma_f32_16x16x32_f16(false, A.v, false, Bop[c].v,
                                                   (short)0, acc, false, false);
      // C layout: VGPR r = row M=r (lanes 0-15) / M=r+8 (lanes 16-31), col=lrow
#pragma unroll
      for (int r = 0; r < 8; ++r) {
        int jr = jt*16 + r + (lo ? 0 : 8);
        float val = silu_f(acc[r]);
        sum[c] += (jr == i) ? 0.f : val;   // diagonal mask (j == i)
      }
    }
  }
  // combine the two half-wave partial row sums, store 128-wide s row
#pragma unroll
  for (int c = 0; c < 8; ++c) {
    float v2 = sum[c] + __shfl_xor(sum[c], 16, 32);
    if (lo) s[(size_t)wid * H + c*16 + lrow] = v2;
  }
}

// build 4 K=32 f16 A-chunks from a 16x128 f32 tile in LDS (C-layout staging)
__device__ __forceinline__ void load_A_from_lds(const float* L, int lrow, bool lo,
                                                V16H A[4]) {
#pragma unroll
  for (int kc = 0; kc < 4; ++kc) {
    const float* p = L + lrow*LDS_STRIDE + kc*32 + (lo ? 0 : 8);
#pragma unroll
    for (int e = 0; e < 16; ++e) A[kc].h[e] = (_Float16)p[e < 8 ? e : e + 8];
  }
}

// ---------------------------------------------------------------------------
// Phase 2: per 16-row i-tile:
//   m  = s @ phi_w1 + 63*phi_b1                       (32 WMMA)
//   h0 = silu([x | m] @ psi_w0 + psi_b0)              (40 WMMA)
//   h1 = silu(h0 @ psi_w1 + psi_b1)                   (32 WMMA)
//   dx = tanh(h1 @ psi_w2 + psi_b2) * softplus(bf)    ( 4 WMMA)
// One wave per i-tile, 4 waves (one full b) per block.
// ---------------------------------------------------------------------------
__global__ __launch_bounds__(128) void phase2(
    const float* __restrict__ x, const float* __restrict__ s,
    const unsigned int* __restrict__ w1p,  const float* __restrict__ phi_b1,
    const unsigned int* __restrict__ w0xp, const unsigned int* __restrict__ w0mp,
    const float* __restrict__ psi_b0,
    const unsigned int* __restrict__ pw1p, const float* __restrict__ psi_b1,
    const unsigned int* __restrict__ pw2p, const float* __restrict__ psi_b2,
    const float* __restrict__ bf_raw, float* __restrict__ out) {
  __shared__ float lds[4][16 * LDS_STRIDE];
  int wave = threadIdx.x >> 5;
  int lane = threadIdx.x & 31;
  int b = blockIdx.x;
  int ibase = wave * 16;
  int lrow = lane & 15;
  bool lo = lane < 16;
  float* L = lds[wave];

  // ---- A chunks from s (this lane's row = ibase + lrow), f32 -> f16
  V16H sA[4];
  const float* srow = s + ((size_t)(b*NPART + ibase + lrow)) * H + (lo ? 0 : 8);
#pragma unroll
  for (int kc = 0; kc < 4; ++kc) {
    const float* p = srow + kc*32;
#pragma unroll
    for (int e = 0; e < 16; ++e) sA[kc].h[e] = (_Float16)p[e < 8 ? e : e + 8];
  }

  // ---- m = s @ W1 + 63*b1, staged into LDS tile (C-layout scatter)
#pragma unroll
  for (int c = 0; c < 8; ++c) {
    v8f_t acc;
    float bb = 63.0f * phi_b1[c*16 + lrow];
#pragma unroll
    for (int r = 0; r < 8; ++r) acc[r] = bb;
#pragma unroll
    for (int kc = 0; kc < 4; ++kc) {
      V16H Bc;
      const unsigned int* p = w1p + ((size_t)((kc*8 + c)*32 + lane)) * 8;
#pragma unroll
      for (int vv = 0; vv < 8; ++vv) Bc.u[vv] = p[vv];
      acc = __builtin_amdgcn_wmma_f32_16x16x32_f16(false, sA[kc].v, false, Bc.v,
                                                   (short)0, acc, false, false);
    }
#pragma unroll
    for (int r = 0; r < 8; ++r)
      L[(r + (lo?0:8)) * LDS_STRIDE + c*16 + lrow] = acc[r];
  }

  // ---- x-feature chunk (K=3 padded to 32) for psi layer 0
  const float* xr = x + ((size_t)(b*NPART + ibase + lrow)) * DIM;
  V16H Ax;
#pragma unroll
  for (int e = 0; e < 16; ++e) Ax.h[e] = (_Float16)0.0f;
  if (lo) { Ax.h[0]=(_Float16)xr[0]; Ax.h[1]=(_Float16)xr[1]; Ax.h[2]=(_Float16)xr[2]; }

  // ---- psi layer 0: h0 = silu(x@W0x + m@W0m + b0)
  V16H Am[4];
  load_A_from_lds(L, lrow, lo, Am);          // read m before overwriting LDS
#pragma unroll
  for (int c = 0; c < 8; ++c) {
    v8f_t acc;
    float bb = psi_b0[c*16 + lrow];
#pragma unroll
    for (int r = 0; r < 8; ++r) acc[r] = bb;
    {
      V16H Bc;
      const unsigned int* p = w0xp + ((size_t)(c*32 + lane)) * 8;
#pragma unroll
      for (int vv = 0; vv < 8; ++vv) Bc.u[vv] = p[vv];
      acc = __builtin_amdgcn_wmma_f32_16x16x32_f16(false, Ax.v, false, Bc.v,
                                                   (short)0, acc, false, false);
    }
#pragma unroll
    for (int kc = 0; kc < 4; ++kc) {
      V16H Bc;
      const unsigned int* p = w0mp + ((size_t)((kc*8 + c)*32 + lane)) * 8;
#pragma unroll
      for (int vv = 0; vv < 8; ++vv) Bc.u[vv] = p[vv];
      acc = __builtin_amdgcn_wmma_f32_16x16x32_f16(false, Am[kc].v, false, Bc.v,
                                                   (short)0, acc, false, false);
    }
#pragma unroll
    for (int r = 0; r < 8; ++r)
      L[(r + (lo?0:8)) * LDS_STRIDE + c*16 + lrow] = silu_f(acc[r]);
  }

  // ---- psi layer 1: h1 = silu(h0 @ W1 + b1)
  V16H Ah[4];
  load_A_from_lds(L, lrow, lo, Ah);
#pragma unroll
  for (int c = 0; c < 8; ++c) {
    v8f_t acc;
    float bb = psi_b1[c*16 + lrow];
#pragma unroll
    for (int r = 0; r < 8; ++r) acc[r] = bb;
#pragma unroll
    for (int kc = 0; kc < 4; ++kc) {
      V16H Bc;
      const unsigned int* p = pw1p + ((size_t)((kc*8 + c)*32 + lane)) * 8;
#pragma unroll
      for (int vv = 0; vv < 8; ++vv) Bc.u[vv] = p[vv];
      acc = __builtin_amdgcn_wmma_f32_16x16x32_f16(false, Ah[kc].v, false, Bc.v,
                                                   (short)0, acc, false, false);
    }
#pragma unroll
    for (int r = 0; r < 8; ++r)
      L[(r + (lo?0:8)) * LDS_STRIDE + c*16 + lrow] = silu_f(acc[r]);
  }

  // ---- psi layer 2 (128 -> 3, one col tile) + tanh + softplus scale
  V16H A2[4];
  load_A_from_lds(L, lrow, lo, A2);
  v8f_t acc;
  float bb = (lrow < 3) ? psi_b2[lrow] : 0.f;
#pragma unroll
  for (int r = 0; r < 8; ++r) acc[r] = bb;
#pragma unroll
  for (int kc = 0; kc < 4; ++kc) {
    V16H Bc;
    const unsigned int* p = pw2p + ((size_t)(kc*32 + lane)) * 8;
#pragma unroll
    for (int vv = 0; vv < 8; ++vv) Bc.u[vv] = p[vv];
    acc = __builtin_amdgcn_wmma_f32_16x16x32_f16(false, A2[kc].v, false, Bc.v,
                                                 (short)0, acc, false, false);
  }
  // softplus(bf_scale_raw) = log2(1+exp(bf)) * ln2  (once per wave, fast path)
  float sp = __builtin_amdgcn_logf(1.0f + fast_exp(bf_raw[0])) * 0.6931471805599453f;
  if (lrow < 3) {
#pragma unroll
    for (int r = 0; r < 8; ++r) {
      int row = ibase + r + (lo ? 0 : 8);
      out[((size_t)b*NPART + row)*DIM + lrow] = fast_tanh(acc[r]) * sp;
    }
  }
}

// ---------------------------------------------------------------------------
extern "C" void kernel_launch(void* const* d_in, const int* in_sizes, int n_in,
                              void* d_out, int out_size, void* d_ws, size_t ws_size,
                              hipStream_t stream) {
  (void)in_sizes; (void)n_in; (void)out_size; (void)ws_size;
  const float* x      = (const float*)d_in[0];
  // d_in[1] = spin: unused (weight is all-ones in the reference)
  const float* phi_w0 = (const float*)d_in[2];
  const float* phi_b0 = (const float*)d_in[3];
  const float* phi_w1 = (const float*)d_in[4];
  const float* phi_b1 = (const float*)d_in[5];
  const float* psi_w0 = (const float*)d_in[6];
  const float* psi_b0 = (const float*)d_in[7];
  const float* psi_w1 = (const float*)d_in[8];
  const float* psi_b1 = (const float*)d_in[9];
  const float* psi_w2 = (const float*)d_in[10];
  const float* psi_b2 = (const float*)d_in[11];
  const float* bf_raw = (const float*)d_in[12];

  // workspace layout (bytes): packed f16 weights, then 16MB s buffer
  char* ws = (char*)d_ws;
  unsigned int* w0p  = (unsigned int*)(ws + 0);        //  2048 dw
  unsigned int* w1p  = (unsigned int*)(ws + 8192);     //  8192 dw
  unsigned int* w0xp = (unsigned int*)(ws + 40960);    //  2048 dw
  unsigned int* w0mp = (unsigned int*)(ws + 49152);    //  8192 dw
  unsigned int* pw1p = (unsigned int*)(ws + 81920);    //  8192 dw
  unsigned int* pw2p = (unsigned int*)(ws + 114688);   //  1024 dw
  float* s = (float*)(ws + (1 << 17));                 // B*N*128 f32 = 16 MB

  // weight repack (tiny; runs once per launch, stream-ordered)
  prep_weights<<<   8, 256, 0, stream>>>(phi_w0, w0p,  0,  11, 128, 128, 8, 1);
  prep_weights<<<  32, 256, 0, stream>>>(phi_w1, w1p,  0, 128, 128, 128, 8, 4);
  prep_weights<<<   8, 256, 0, stream>>>(psi_w0, w0xp, 0,   3, 128, 128, 8, 1);
  prep_weights<<<  32, 256, 0, stream>>>(psi_w0, w0mp, 3, 128, 128, 128, 8, 4);
  prep_weights<<<  32, 256, 0, stream>>>(psi_w1, pw1p, 0, 128, 128, 128, 8, 4);
  prep_weights<<<   4, 256, 0, stream>>>(psi_w2, pw2p, 0, 128,   3,   3, 1, 4);

  phase1<<<BATCH*NPART/8, 256, 0, stream>>>(x, w0p, phi_b0, s);
  phase2<<<BATCH, 128, 0, stream>>>(x, s, w1p, phi_b1, w0xp, w0mp, psi_b0,
                                    pw1p, psi_b1, pw2p, psi_b2, bf_raw,
                                    (float*)d_out);
}
